// NTXentLoss_42494406427305
// MI455X (gfx1250) — matrix operand order repaired and checked by reading, not compiled, not run on hardware
//
#include <hip/hip_runtime.h>
#include <math.h>

typedef __attribute__((ext_vector_type(16))) _Float16 v16h;
typedef __attribute__((ext_vector_type(8)))  _Float16 v8h;
typedef __attribute__((ext_vector_type(8)))  float    v8f;
typedef __attribute__((ext_vector_type(4)))  float    v4f;

#define N_ROWS 4096
#define TWO_N  8192
#define DIM    256
#define EPSN   1e-8f
#define LDS_STRIDE 264          // 256 + 8 halves pad -> rotates LDS banks
#define COLP   4                // column partitions (parallelism)
#define TILES_PER_PART (TWO_N / 16 / COLP)   // 128
#define LOG2E  1.44269504088896340736f
#define LN2    0.69314718055994530942f

// --------------------------------------------- init d_out + per-row partials
__global__ __launch_bounds__(256) void ntxent_init(
    float* __restrict__ out, float* __restrict__ srow,
    float* __restrict__ bvrow, float* __restrict__ tgrow) {
    const int i = (int)(blockIdx.x * blockDim.x + threadIdx.x);
    if (i < TWO_N) {
        srow[i]  = 0.0f;
        bvrow[i] = -__builtin_inff();
        tgrow[i] = -__builtin_inff();
    }
    if (i < 2) out[i] = 0.0f;
}

// ------------------------------------------------- normalize rows -> f16 zn
__global__ __launch_bounds__(256) void ntxent_normalize(
    const float* __restrict__ z1, const float* __restrict__ z2,
    _Float16* __restrict__ zn) {
    const int row  = (int)((blockIdx.x * blockDim.x + threadIdx.x) >> 5);
    const int lane = (int)(threadIdx.x & 31);
    if (row >= TWO_N) return;
    const float* src = (row < N_ROWS) ? (z1 + (size_t)row * DIM)
                                      : (z2 + (size_t)(row - N_ROWS) * DIM);
    v4f a = *(const v4f*)(src + lane * 8);
    v4f b = *(const v4f*)(src + lane * 8 + 4);
    float ss = 0.0f;
#pragma unroll
    for (int i = 0; i < 4; ++i) { ss += a[i] * a[i]; ss += b[i] * b[i]; }
#pragma unroll
    for (int m = 16; m > 0; m >>= 1) ss += __shfl_xor(ss, m, 32);
    const float inv = 1.0f / fmaxf(sqrtf(ss), EPSN);
    v8h h;
#pragma unroll
    for (int i = 0; i < 4; ++i) {
        h[i]     = (_Float16)(a[i] * inv);
        h[i + 4] = (_Float16)(b[i] * inv);
    }
    *(v8h*)(zn + (size_t)row * DIM + lane * 8) = h;
}

// --------------- fused Gram-WMMA + bounded-softmax partials (log2 domain)
// A fragments pre-scaled by (1/temp)*log2(e): WMMA emits logits*log2(e), so
// the softmax sum is a bare v_exp_f32 (exp2) per element.
__global__ __launch_bounds__(256) void ntxent_main(
    const _Float16* __restrict__ zn, const float* __restrict__ logtemp,
    float* __restrict__ srow, float* __restrict__ bvrow,
    float* __restrict__ tgrow) {
    __shared__ _Float16 bpan[2][16 * LDS_STRIDE];   // double-buffered B panel

    const int tid  = (int)threadIdx.x;
    const int wave = tid >> 5;
    const int lane = tid & 31;
    const int hi   = lane >> 4;
    const int ln   = lane & 15;
    const int rowbase = ((int)blockIdx.x * 8 + wave) * 16;
    // wave-uniform: force into SGPRs so tile checks are s_cmp/s_cbranch
    const int dct = __builtin_amdgcn_readfirstlane(rowbase >> 4);
    const int tct = __builtin_amdgcn_readfirstlane(
                        ((rowbase + N_ROWS) & (TWO_N - 1)) >> 4);
    const float invT = __expf(-logtemp[0]);

    // ---- Preload A fragments (16 rows x K=256), fold invT*log2e into A.
    v16h afrag[8];
    {
        const _Float16* arow = zn + (size_t)(rowbase + ln) * DIM + hi * 16;
        const _Float16 sT = (_Float16)(invT * LOG2E);
#pragma unroll
        for (int kc = 0; kc < 8; ++kc) {
            union { v16h v; v8h h[2]; } a;
            a.h[0] = *(const v8h*)(arow + kc * 32);
            a.h[1] = *(const v8h*)(arow + kc * 32 + 8);
            afrag[kc] = a.v * sT;     // v_pk_mul_f16, once
        }
    }

    float s[8], bv[8], tg[8];
#pragma unroll
    for (int r = 0; r < 8; ++r) {
        s[r] = 0.0f; bv[r] = -__builtin_inff(); tg[r] = -__builtin_inff();
    }

    const int crow = tid >> 4;          // cooperative staging coords
    const int ccol = (tid & 15) * 16;
    const int ct0 = (int)blockIdx.y * TILES_PER_PART;
    const int ct1 = ct0 + TILES_PER_PART;

    // stage first panel into buffer 0
    {
        const _Float16* g = zn + (size_t)(ct0 * 16 + crow) * DIM + ccol;
        v8h p0 = *(const v8h*)(g);
        v8h p1 = *(const v8h*)(g + 8);
        _Float16* l = bpan[0] + crow * LDS_STRIDE + ccol;
        *(v8h*)(l) = p0; *(v8h*)(l + 8) = p1;
    }

    int p = 0;
    for (int ct = ct0; ct < ct1; ++ct) {
        __syncthreads();    // buf[p] ready; buf[p^1] free to overwrite
        if (ct + 1 < ct1) { // prefetch next panel into the other buffer
            const _Float16* g = zn + (size_t)((ct + 1) * 16 + crow) * DIM + ccol;
            v8h p0 = *(const v8h*)(g);
            v8h p1 = *(const v8h*)(g + 8);
            _Float16* l = bpan[p ^ 1] + crow * LDS_STRIDE + ccol;
            *(v8h*)(l) = p0; *(v8h*)(l + 8) = p1;
        }

        // ---- 16x16 tile: software-pipelined B fragments (2 register sets)
        // so ds_load_b128 of chunk kc+1 overlaps v_wmma of chunk kc.
        const _Float16* lbase = bpan[p] + ln * LDS_STRIDE + hi * 16;
        union Bf { v16h v; v8h h[2]; };
        Bf b[2];
        b[0].h[0] = *(const v8h*)(lbase);
        b[0].h[1] = *(const v8h*)(lbase + 8);
        v8f c = {};
#pragma unroll
        for (int kc = 0; kc < 8; ++kc) {
            if (kc < 7) {
                b[(kc + 1) & 1].h[0] = *(const v8h*)(lbase + (kc + 1) * 32);
                b[(kc + 1) & 1].h[1] = *(const v8h*)(lbase + (kc + 1) * 32 + 8);
            }
            c = __builtin_amdgcn_wmma_f32_16x16x32_f16(
                    false, afrag[kc], false, b[kc & 1].v, (short)0, c,
                    false, false);
        }

        // ---- rare, now-scalar branches: target capture, diagonal mask
        if (ct == tct) {
#pragma unroll
            for (int r = 0; r < 8; ++r)
                if (ln == r + 8 * hi) tg[r] = c[r];
        }
        if (ct == dct) {
#pragma unroll
            for (int r = 0; r < 8; ++r)
                if (ln == r + 8 * hi) c[r] = -__builtin_inff();  // exp2 -> 0
        }

        // ---- hot epilogue: 3 VALU per element (v_exp co-executes w/ WMMA)
#pragma unroll
        for (int r = 0; r < 8; ++r) {
            s[r]  += __builtin_exp2f(c[r]);
            bv[r]  = fmaxf(bv[r], c[r]);
        }
        p ^= 1;
    }

    // ---- butterfly over the 16 lanes sharing each row
#pragma unroll
    for (int mask = 1; mask < 16; mask <<= 1) {
#pragma unroll
        for (int r = 0; r < 8; ++r) {
            s[r]  += __shfl_xor(s[r], mask, 32);
            bv[r]  = fmaxf(bv[r], __shfl_xor(bv[r], mask, 32));
            tg[r]  = fmaxf(tg[r], __shfl_xor(tg[r], mask, 32));
        }
    }

    // ---- commit per-row partials (combine across column partitions)
    if (ln == 0) {
#pragma unroll
        for (int r = 0; r < 8; ++r) {
            const int i = rowbase + r + 8 * hi;
            atomicAdd(&srow[i], s[r]);
            atomicMax(&bvrow[i], bv[r]);
            atomicMax(&tgrow[i], tg[r]);
        }
    }
}

// ----------------------------------------------- per-row -> (loss, correct)
// tg/bv are logits*log2(e); convert back with ln2 for the loss.
__global__ __launch_bounds__(256) void ntxent_finalize(
    const float* __restrict__ srow, const float* __restrict__ bvrow,
    const float* __restrict__ tgrow, float* __restrict__ out) {
    __shared__ float L, C;
    const int tid = (int)threadIdx.x;
    const int i = (int)blockIdx.x * 256 + tid;
    if (tid == 0) { L = 0.0f; C = 0.0f; }
    __syncthreads();
    const float tg = tgrow[i];
    const float l  = __logf(srow[i]) - tg * LN2;    // -log p[target]
    const float c  = (bvrow[i] == tg) ? 1.0f : 0.0f;
    atomicAdd(&L, l);
    atomicAdd(&C, c);
    __syncthreads();
    if (tid == 0) {
        atomicAdd(&out[0], L * (1.0f / (float)TWO_N));
        atomicAdd(&out[1], C * 0.5f);
    }
}

// --------------------------------------------------------------------------
extern "C" void kernel_launch(void* const* d_in, const int* in_sizes, int n_in,
                              void* d_out, int out_size, void* d_ws, size_t ws_size,
                              hipStream_t stream) {
    const float* z1 = (const float*)d_in[0];
    const float* z2 = (const float*)d_in[1];
    const float* lt = (const float*)d_in[2];
    float* out = (float*)d_out;

    char* ws = (char*)d_ws;
    _Float16* zn = (_Float16*)ws;                            // 4 MB
    float* srow  = (float*)(ws + (size_t)TWO_N * DIM * 2);   // 32 KB
    float* bvrow = srow + TWO_N;                             // 32 KB
    float* tgrow = bvrow + TWO_N;                            // 32 KB

    ntxent_init<<<TWO_N / 256, 256, 0, stream>>>(out, srow, bvrow, tgrow);
    ntxent_normalize<<<TWO_N / 8, 256, 0, stream>>>(z1, z2, zn);
    ntxent_main<<<dim3(TWO_N / (16 * 8), COLP), 256, 0, stream>>>(
        zn, lt, srow, bvrow, tgrow);
    ntxent_finalize<<<TWO_N / 256, 256, 0, stream>>>(srow, bvrow, tgrow, out);
}
